// LSTMFrame_2035814499069
// MI455X (gfx1250) — compile-verified, matrix operand order checked by reading, not run on hardware
//
#include <hip/hip_runtime.h>
#include <hip/hip_bf16.h>

typedef __attribute__((ext_vector_type(16))) _Float16 v16h;
typedef __attribute__((ext_vector_type(8)))  _Float16 v8h;
typedef __attribute__((ext_vector_type(4)))  _Float16 v4h;
typedef __attribute__((ext_vector_type(8)))  float    v8f;

#define Tn 512
#define Bn 64
#define In 512
#define Hn 512
#define NWG 64   // 32 WGs per direction (16 h-cols each) x 2 directions

__device__ __forceinline__ float sigm_f(float x) {
    return 1.0f / (1.0f + __expf(-x));
}
__device__ __forceinline__ float tanh_f(float x) {
    // stable: large +x -> 1, large -x -> -1
    return 1.0f - 2.0f / (__expf(2.0f * x) + 1.0f);
}

union ABf { v16h v; v8h h[2]; };

// ---------------- fp32 -> fp16 convert (x input) ----------------
__global__ void f32_to_f16_kernel(const float* __restrict__ in,
                                  _Float16* __restrict__ out, int n4) {
    int i = blockIdx.x * blockDim.x + threadIdx.x;
    if (i < n4) {
        float4 v = ((const float4*)in)[i];
        v4h h = { (_Float16)v.x, (_Float16)v.y, (_Float16)v.z, (_Float16)v.w };
        ((v4h*)out)[i] = h;
    }
}

// ---------------- persistent bidirectional LSTM layer ----------------
// grid = 64 WGs: dir = blockIdx.x/32, 16 h-columns per WG. block = 128 (4 waves).
// Wave m computes gate tiles (i,f,g,o) for batch rows [16m,16m+16) and the WG's
// 16 columns, K = K_IN (x part) + 512 (h part), via v_wmma_f32_16x16x32_f16.
template<int K_IN, bool L0>
__global__ __launch_bounds__(128, 1)
void lstm_layer_kernel(const _Float16* __restrict__ xf16,   // (T,B,K_IN) f16
                       const float* __restrict__ Wih,       // (2,2048,K_IN)
                       const float* __restrict__ Whh,       // (2,2048,512)
                       const float* __restrict__ bih,       // (2,2048)
                       const float* __restrict__ bhh,       // (2,2048)
                       _Float16* __restrict__ hbuf,         // (2 dir,2 parity,B*H) f16
                       _Float16* __restrict__ hsF16,        // (T,B,2H) f16   (L0)
                       float* __restrict__ outF32,          // (T,B,2H) f32   (L1)
                       float* __restrict__ hn,              // layer h_n base (+dir*B*H)
                       float* __restrict__ cn,              // layer c_n base (+dir*B*H)
                       unsigned* __restrict__ ctr) {
    constexpr int Kt   = K_IN + Hn;
    constexpr int LROW = Kt + 8;            // +16B pad: bank-conflict-free ds_load_b128
    __shared__ __align__(16) _Float16 smem[64 * LROW];

    const int tid  = threadIdx.x;
    const int wg   = blockIdx.x;
    const int dir  = wg >> 5;               // 0 = fwd, 1 = bwd
    const int c0   = (wg & 31) * 16;        // this WG's first h-column
    const int m    = tid >> 5;              // wave id = M tile
    const int lane = tid & 31;
    const int l16  = lane & 15;
    const int kh   = lane >> 4;

    // ---- preload [Wih|Whh] slice into LDS as f16 (64 rows = 4 gates x 16 cols) ----
    constexpr int row4 = Kt / 4;
    for (int idx = tid; idx < 64 * row4; idx += 128) {
        int r  = idx / row4;
        int k4 = (idx - r * row4) * 4;
        int wr = (r >> 4) * 512 + c0 + (r & 15);          // row within 4H gate block
        const float* src = (k4 < K_IN)
            ? (Wih + (size_t)(dir * 2048 + wr) * K_IN + k4)
            : (Whh + (size_t)(dir * 2048 + wr) * Hn + (k4 - K_IN));
        float4 w = *(const float4*)src;
        v4h h4 = { (_Float16)w.x, (_Float16)w.y, (_Float16)w.z, (_Float16)w.w };
        *(v4h*)(&smem[r * LROW + k4]) = h4;
    }

    // per-lane gate bias (N = l16), broadcast over the 8 accumulator rows
    float bias4[4];
    #pragma unroll
    for (int g = 0; g < 4; ++g) {
        int gc = dir * 2048 + g * 512 + c0 + l16;
        bias4[g] = bih[gc] + bhh[gc];
    }

    float creg[8];                                        // cell state, 16x16 tile slice
    #pragma unroll
    for (int v = 0; v < 8; ++v) creg[v] = 0.0f;

    __syncthreads();

    const int arow = 16 * m + l16;                        // A-matrix row = batch index
    const int hcol = c0 + l16;                            // output column (N = l16)

    for (int s = 0; s < Tn; ++s) {
        const int t = dir ? (Tn - 1 - s) : s;
        const _Float16* hprev = hbuf + (size_t)(dir * 2 + (s & 1)) * (Bn * Hn);
        _Float16*       hnext = hbuf + (size_t)(dir * 2 + ((s + 1) & 1)) * (Bn * Hn);
        const _Float16* xrow  = xf16 + (size_t)t * Bn * K_IN;

        v8f acc[4];
        #pragma unroll
        for (int g = 0; g < 4; ++g)
            #pragma unroll
            for (int v = 0; v < 8; ++v) acc[g][v] = bias4[g];

        #pragma unroll 4
        for (int ko = 0; ko < Kt; ko += 32) {
            // A fragment (16x32 f16): two b128 loads per lane, row-major source
            const _Float16* ab = (ko < K_IN)
                ? (xrow + (size_t)arow * K_IN + ko + kh * 8)
                : (hprev + (size_t)arow * Hn + (ko - K_IN) + kh * 8);
            ABf a;
            a.h[0] = *(const v8h*)(ab);
            a.h[1] = *(const v8h*)(ab + 16);
            // B fragments for the 4 gates from LDS-resident transposed weights
            #pragma unroll
            for (int g = 0; g < 4; ++g) {
                const _Float16* bp = &smem[(g * 16 + l16) * LROW + ko + kh * 8];
                ABf b;
                b.h[0] = *(const v8h*)(bp);
                b.h[1] = *(const v8h*)(bp + 16);
                acc[g] = __builtin_amdgcn_wmma_f32_16x16x32_f16(
                    false, a.v, false, b.v, (short)0, acc[g], false, false);
            }
        }

        // ---- lane-local LSTM cell update (accumulator layout: M = 8*kh+v, N = l16) ----
        #pragma unroll
        for (int v = 0; v < 8; ++v) {
            float gi = acc[0][v], gf = acc[1][v], gg = acc[2][v], go = acc[3][v];
            float c = sigm_f(gf) * creg[v] + sigm_f(gi) * tanh_f(gg);
            float h = sigm_f(go) * tanh_f(c);
            creg[v] = c;
            int batch = 16 * m + 8 * kh + v;
            hnext[(size_t)batch * Hn + hcol] = (_Float16)h;
            if constexpr (L0) {
                hsF16[((size_t)t * Bn + batch) * (2 * Hn) + dir * Hn + hcol] = (_Float16)h;
            } else {
                outF32[((size_t)t * Bn + batch) * (2 * Hn) + dir * Hn + hcol] = h;
            }
            if (s == Tn - 1) {
                hn[(size_t)dir * Bn * Hn + (size_t)batch * Hn + hcol] = h;
                cn[(size_t)dir * Bn * Hn + (size_t)batch * Hn + hcol] = c;
            }
        }

        // ---- device-scope split barrier between timesteps ----
        if (s + 1 < Tn) {
            __threadfence();
            __syncthreads();
            if (tid == 0) {
                __hip_atomic_fetch_add(ctr, 1u, __ATOMIC_RELEASE, __HIP_MEMORY_SCOPE_AGENT);
                unsigned tgt = (unsigned)NWG * (unsigned)(s + 1);
                while (__hip_atomic_load(ctr, __ATOMIC_ACQUIRE, __HIP_MEMORY_SCOPE_AGENT) < tgt)
                    __builtin_amdgcn_s_sleep(2);
            }
            __syncthreads();
        }
    }
}

// ---------------- host launcher ----------------
extern "C" void kernel_launch(void* const* d_in, const int* in_sizes, int n_in,
                              void* d_out, int out_size, void* d_ws, size_t ws_size,
                              hipStream_t stream) {
    const float* x    = (const float*)d_in[0];
    const float* Wih0 = (const float*)d_in[1];
    const float* Whh0 = (const float*)d_in[2];
    const float* bih0 = (const float*)d_in[3];
    const float* bhh0 = (const float*)d_in[4];
    const float* Wih1 = (const float*)d_in[5];
    const float* Whh1 = (const float*)d_in[6];
    const float* bih1 = (const float*)d_in[7];
    const float* bhh1 = (const float*)d_in[8];
    float* out = (float*)d_out;

    // workspace layout
    char* ws = (char*)d_ws;
    const size_t xF16_bytes = (size_t)Tn * Bn * In * 2;          //  32 MB
    const size_t hs0_bytes  = (size_t)Tn * Bn * 2 * Hn * 2;      //  64 MB
    const size_t hbuf_bytes = (size_t)2 * 2 * Bn * Hn * 2;       // 256 KB
    _Float16* xF16 = (_Float16*)ws;
    _Float16* hs0  = (_Float16*)(ws + xF16_bytes);
    _Float16* hbuf = (_Float16*)(ws + xF16_bytes + hs0_bytes);
    unsigned* ctr  = (unsigned*)(ws + xF16_bytes + hs0_bytes + hbuf_bytes);

    const size_t TB2H = (size_t)Tn * Bn * 2 * Hn;                // layer_output elems
    const size_t BH   = (size_t)Bn * Hn;

    // zero h ping-pong buffers + barrier counter
    hipMemsetAsync(hbuf, 0, hbuf_bytes + 64, stream);

    // convert x to f16
    int n4 = Tn * Bn * In / 4;
    f32_to_f16_kernel<<<(n4 + 255) / 256, 256, 0, stream>>>(x, xF16, n4);

    // layer 0 (bidirectional, persistent): writes hs0 (f16) + h_n/c_n slots 0,1
    lstm_layer_kernel<In, true><<<NWG, 128, 0, stream>>>(
        xF16, Wih0, Whh0, bih0, bhh0, hbuf, hs0, nullptr,
        out + TB2H, out + TB2H + 4 * BH, ctr);

    // reset barrier + h buffers for layer 1
    hipMemsetAsync(hbuf, 0, hbuf_bytes + 64, stream);

    // layer 1: input = hs0 (T,B,1024) f16, writes layer_output f32 + h_n/c_n slots 2,3
    lstm_layer_kernel<2 * Hn, false><<<NWG, 128, 0, stream>>>(
        hs0, Wih1, Whh1, bih1, bhh1, hbuf, nullptr, out,
        out + TB2H + 2 * BH, out + TB2H + 6 * BH, ctr);
}
